// TemporalEncoder_52587579572481
// MI455X (gfx1250) — matrix operand order repaired and checked by reading, not compiled, not run on hardware
//
#include <hip/hip_runtime.h>

typedef __attribute__((ext_vector_type(16))) _Float16 v16h;
typedef __attribute__((ext_vector_type(8)))  _Float16 v8h;
typedef __attribute__((ext_vector_type(8)))  float    v8f;
typedef unsigned int v4u  __attribute__((ext_vector_type(4)));
typedef int          v4i  __attribute__((ext_vector_type(4)));
typedef int          v8i  __attribute__((ext_vector_type(8)));

#define WMMA_F16(a,b,c) __builtin_amdgcn_wmma_f32_16x16x32_f16(false,(a),false,(b),(short)0,(c),false,false)

static constexpr int D_IN = 256, Hd = 512, NH = 8, Bb = 8, Tt = 1024;
static constexpr int HD = 64;             // head dim
static constexpr int ROWS = Bb * Tt;      // 8192

#define GF_PE 1
#define GF_VT 2

// ---------------- WMMA fragment loaders (ISA 7.12.2 layouts) ----------------

// A fragment: 16x32 f16, rows (row0..row0+15) of row-major src with leading dim lda.
__device__ __forceinline__ v16h load_frag_a(const _Float16* src, int lda, int row0, int k0) {
  int lane = threadIdx.x & 31;
  int m = lane & 15;
  int koff = (lane >> 4) ? 8 : 0;
  const _Float16* p = src + (size_t)(row0 + m) * lda + k0 + koff;
  v8h lo = *(const v8h*)(p);
  v8h hi = *(const v8h*)(p + 16);
  v16h r;
#pragma unroll
  for (int i = 0; i < 8; ++i) { r[i] = lo[i]; r[i + 8] = hi[i]; }
  return r;
}

// B fragment: 32x16, where B[k][n] = W[n0+n][k0+k], W row-major [N,K] leading dim ldw.
__device__ __forceinline__ v16h load_frag_b(const _Float16* W, int ldw, int n0, int k0) {
  int lane = threadIdx.x & 31;
  int n = lane & 15;
  int koff = (lane >> 4) ? 8 : 0;
  const _Float16* p = W + (size_t)(n0 + n) * ldw + k0 + koff;
  v8h lo = *(const v8h*)(p);
  v8h hi = *(const v8h*)(p + 16);
  v16h r;
#pragma unroll
  for (int i = 0; i < 8; ++i) { r[i] = lo[i]; r[i + 8] = hi[i]; }
  return r;
}

// ---------------- TDM: async-load a [rows x K] f16 panel into LDS ----------------
// D# built per ISA 8.3/8.4: 2-D tensor, data_size=2B, tile_dim0=K elems, tile_dim1=rows.
__device__ __forceinline__ void tdm_load_panel(const _Float16* gsrc, unsigned lds_byte_addr,
                                               int K, int rows) {
  unsigned long long ga = (unsigned long long)(size_t)gsrc;
  v4u g0;
  g0[0] = 1u;                                  // count=1 (valid), load, no gather
  g0[1] = lds_byte_addr;                       // lds_addr [63:32]
  g0[2] = (unsigned)(ga & 0xffffffffu);        // global_addr [95:64]
  g0[3] = (unsigned)((ga >> 32) & 0x01ffffffu) // global_addr [120:96]
          | (2u << 30);                        // type=2 ("image") [127:126]
  unsigned dim0 = (unsigned)K, dim1 = (unsigned)rows;
  unsigned str0 = (unsigned)K;
  v8i g1;
  g1[0] = (int)(1u << 16);                     // data_size=1 (2 bytes) [17:16]
  g1[1] = (int)((dim0 & 0xffffu) << 16);       // tensor_dim0 lo -> [63:48]
  g1[2] = (int)((dim0 >> 16) | ((dim1 & 0xffffu) << 16));   // dim0 hi / dim1 lo
  g1[3] = (int)((dim1 >> 16) | ((unsigned)K << 16));        // dim1 hi / tile_dim0 [127:112]
  g1[4] = (int)((unsigned)rows);               // tile_dim1 [143:128], tile_dim2=0
  g1[5] = (int)str0;                           // tensor_dim0_stride lo [191:160]
  g1[6] = 0;                                   // stride hi / dim1_stride lo
  g1[7] = 0;
  v4i gz = {0, 0, 0, 0};
#if defined(__clang_major__) && (__clang_major__ >= 23)
  v8i gz8 = {0, 0, 0, 0, 0, 0, 0, 0};
  __builtin_amdgcn_tensor_load_to_lds(g0, g1, gz, gz, gz8, 0);
#else
  __builtin_amdgcn_tensor_load_to_lds(g0, g1, gz, gz, 0);
#endif
}

// ---------------- f32 -> f16 convert ----------------
__global__ void k_f32_to_f16(const float* __restrict__ src, _Float16* __restrict__ dst, int n) {
  int i = blockIdx.x * 256 + threadIdx.x;
  if (i < n) dst[i] = (_Float16)src[i];
}

// ---------------- epilogue store ----------------
__device__ __forceinline__ void gemm_store(v8f acc, int rowb, int colb,
                                           const float* bias, float* Cf, _Float16* Ch,
                                           int N, int flags, int hi, int ln) {
#pragma unroll
  for (int r = 0; r < 8; ++r) {
    int row = rowb + r + 8 * hi;
    int col = colb + ln;
    float v = acc[r] + (bias ? bias[col] : 0.f);
    if (flags & GF_PE) {
      int t = row & (Tt - 1);
      int dpair = col & ~1;
      float ang = (float)t * __expf((float)dpair * (-9.21034037f / 512.0f));
      v += (col & 1) ? __cosf(ang) : __sinf(ang);
    }
    if (Cf) Cf[(size_t)row * N + col] = v;
    if (Ch) {
      if (flags & GF_VT) {  // scatter to V^T layout: [b*NH+h][d][t]
        int bbi = row >> 10, tti = row & (Tt - 1);
        int hh = col >> 6,  dd = col & (HD - 1);
        Ch[(((size_t)(bbi * NH + hh) * HD) + dd) * Tt + tti] = (_Float16)v;
      } else {
        Ch[(size_t)row * N + col] = (_Float16)v;
      }
    }
  }
}

// ---------------- WMMA GEMM: C[M,N] = A[M,K] @ W[N,K]^T + bias ----------------
// block = 128 threads (4 waves). Block tile 64(M) x 128(N); wave tile 64 x 32
// (4x2 fragments). A-panel (64 x K) staged in LDS once per block by the TDM and
// shared by all 4 waves; W fragments stream from global (L2-resident).
__global__ void k_gemm_wmma(const _Float16* __restrict__ A, const _Float16* __restrict__ W,
                            const float* __restrict__ bias, float* __restrict__ Cf,
                            _Float16* __restrict__ Ch, int M, int N, int K, int flags) {
  __shared__ _Float16 ldsA[64 * 512];
  int lane = threadIdx.x & 31;
  int wid = threadIdx.x >> 5;
  int m0 = blockIdx.y * 64;
  int n0 = blockIdx.x * 128 + wid * 32;

  if (wid == 0) {
    tdm_load_panel(A + (size_t)m0 * K, (unsigned)(size_t)&ldsA[0], K, 64);
    __builtin_amdgcn_s_wait_tensorcnt(0);
  }
  __syncthreads();

  v8f acc[4][2];
#pragma unroll
  for (int i = 0; i < 4; ++i)
#pragma unroll
    for (int j = 0; j < 2; ++j) acc[i][j] = (v8f){};

  for (int k0 = 0; k0 < K; k0 += 32) {
    __builtin_prefetch(W + (size_t)n0 * K + k0 + 64, 0, 0);
    v16h fb0 = load_frag_b(W, K, n0, k0);
    v16h fb1 = load_frag_b(W, K, n0 + 16, k0);
#pragma unroll
    for (int i = 0; i < 4; ++i) {
      v16h fa = load_frag_a(ldsA, K, 16 * i, k0);
      acc[i][0] = WMMA_F16(fa, fb0, acc[i][0]);
      acc[i][1] = WMMA_F16(fa, fb1, acc[i][1]);
    }
  }
  int hi = lane >> 4, ln = lane & 15;
#pragma unroll
  for (int i = 0; i < 4; ++i) {
    gemm_store(acc[i][0], m0 + 16 * i, n0,      bias, Cf, Ch, N, flags, hi, ln);
    gemm_store(acc[i][1], m0 + 16 * i, n0 + 16, bias, Cf, Ch, N, flags, hi, ln);
  }
}

// ---------------- flash causal attention ----------------
// grid = (T/16, B*NH), block = 32 (one wave per 16-query tile of one head)
__global__ void k_flash_attn(const _Float16* __restrict__ Q, const _Float16* __restrict__ Km,
                             const _Float16* __restrict__ Vt, _Float16* __restrict__ O) {
  __shared__ _Float16 ldsP[16 * 32];
  int lane = threadIdx.x & 31;
  int hi = lane >> 4, ln = lane & 15;
  int bh = blockIdx.y;
  int b = bh >> 3, h = bh & (NH - 1);
  int q0 = blockIdx.x * 16;
  const _Float16* Qp = Q  + (size_t)b * Tt * Hd + h * HD;   // [t][d], lda=Hd
  const _Float16* Kp = Km + (size_t)b * Tt * Hd + h * HD;   // [t][d], lda=Hd
  const _Float16* Vp = Vt + (size_t)bh * HD * Tt;           // [d][t], lda=Tt

  v8f o0 = {}, o1 = {}, o2 = {}, o3 = {};
  float mrow[8], lrow[8];
#pragma unroll
  for (int r = 0; r < 8; ++r) { mrow[r] = -3.0e38f; lrow[r] = 0.f; }

  for (int kb = 0; kb <= q0 + 15; kb += 32) {
    v8f s0 = {}, s1 = {};
#pragma unroll
    for (int d0 = 0; d0 < HD; d0 += 32) {
      v16h aq  = load_frag_a(Qp, Hd, q0, d0);
      v16h bk0 = load_frag_b(Kp, Hd, kb, d0);
      v16h bk1 = load_frag_b(Kp, Hd, kb + 16, d0);
      s0 = WMMA_F16(aq, bk0, s0);
      s1 = WMMA_F16(aq, bk1, s1);
    }
    const float scale = 0.125f;  // 64^-0.5
#pragma unroll
    for (int r = 0; r < 8; ++r) {
      int row = q0 + r + 8 * hi;
      float v0 = s0[r] * scale, v1 = s1[r] * scale;
      if (kb + ln > row)      v0 = -3.0e38f;
      if (kb + 16 + ln > row) v1 = -3.0e38f;
      float mx = fmaxf(v0, v1);
      mx = fmaxf(mx, __shfl_xor(mx, 1));
      mx = fmaxf(mx, __shfl_xor(mx, 2));
      mx = fmaxf(mx, __shfl_xor(mx, 4));
      mx = fmaxf(mx, __shfl_xor(mx, 8));
      float mn = fmaxf(mrow[r], mx);
      float em = __expf(mrow[r] - mn);
      float p0 = __expf(v0 - mn);
      float p1 = __expf(v1 - mn);
      float rs = p0 + p1;
      rs += __shfl_xor(rs, 1); rs += __shfl_xor(rs, 2);
      rs += __shfl_xor(rs, 4); rs += __shfl_xor(rs, 8);
      lrow[r] = lrow[r] * em + rs;
      mrow[r] = mn;
      o0[r] *= em; o1[r] *= em; o2[r] *= em; o3[r] *= em;
      ldsP[(r + 8 * hi) * 32 + ln]      = (_Float16)p0;
      ldsP[(r + 8 * hi) * 32 + 16 + ln] = (_Float16)p1;
    }
    __syncthreads();
    v16h ap = load_frag_a(ldsP, 32, 0, 0);
    v16h bv0 = load_frag_b(Vp, Tt, 0,  kb);
    v16h bv1 = load_frag_b(Vp, Tt, 16, kb);
    v16h bv2 = load_frag_b(Vp, Tt, 32, kb);
    v16h bv3 = load_frag_b(Vp, Tt, 48, kb);
    o0 = WMMA_F16(ap, bv0, o0);
    o1 = WMMA_F16(ap, bv1, o1);
    o2 = WMMA_F16(ap, bv2, o2);
    o3 = WMMA_F16(ap, bv3, o3);
    __syncthreads();
  }
#pragma unroll
  for (int r = 0; r < 8; ++r) {
    float inv = 1.f / lrow[r];
    int row = q0 + r + 8 * hi;
    _Float16* dst = O + ((size_t)b * Tt + row) * Hd + h * HD;
    dst[0 * 16 + ln] = (_Float16)(o0[r] * inv);
    dst[1 * 16 + ln] = (_Float16)(o1[r] * inv);
    dst[2 * 16 + ln] = (_Float16)(o2[r] * inv);
    dst[3 * 16 + ln] = (_Float16)(o3[r] * inv);
  }
}

// ---------------- LayerNorm (optionally concat of two halves + residual) ----------------
__global__ void k_layernorm(const float* __restrict__ in0, const float* __restrict__ in1,
                            const float* __restrict__ res, const float* __restrict__ gam,
                            const float* __restrict__ bet, float* __restrict__ out32,
                            _Float16* __restrict__ out16, int w0, int w1) {
  int row = blockIdx.x;
  int Wt = w0 + w1;
  int tid = threadIdx.x;
  int per = Wt >> 8;
  float vals[4];
  float s = 0.f;
  for (int ii = 0; ii < per; ++ii) {
    int j = tid + (ii << 8);
    float v = (j < w0) ? in0[(size_t)row * w0 + j] : in1[(size_t)row * w1 + (j - w0)];
    if (res) v += res[(size_t)row * Wt + j];
    vals[ii] = v;
    s += v;
  }
  __shared__ float red[8];
  for (int o = 16; o >= 1; o >>= 1) s += __shfl_xor(s, o);
  if ((tid & 31) == 0) red[tid >> 5] = s;
  __syncthreads();
  float tot = 0.f;
  for (int k2 = 0; k2 < 8; ++k2) tot += red[k2];
  float mean = tot / (float)Wt;
  __syncthreads();
  float vs = 0.f;
  for (int ii = 0; ii < per; ++ii) { float d = vals[ii] - mean; vs += d * d; }
  for (int o = 16; o >= 1; o >>= 1) vs += __shfl_xor(vs, o);
  if ((tid & 31) == 0) red[tid >> 5] = vs;
  __syncthreads();
  float vtot = 0.f;
  for (int k2 = 0; k2 < 8; ++k2) vtot += red[k2];
  float inv = rsqrtf(vtot / (float)Wt + 1e-5f);
  for (int ii = 0; ii < per; ++ii) {
    int j = tid + (ii << 8);
    float y = (vals[ii] - mean) * inv * gam[j] + bet[j];
    if (out32) out32[(size_t)row * Wt + j] = y;
    if (out16) out16[(size_t)row * Wt + j] = (_Float16)y;
  }
}

// ---------------- persistent bidirectional LSTM ----------------
// grid = 2 (dir), block = 1024 (32 waves). Each wave owns 16 hidden units and
// computes all four gate tiles for them each step (A fragment from LDS h reused).
__global__ void k_lstm(const float* __restrict__ xg_f, const float* __restrict__ xg_b,
                       const _Float16* __restrict__ whh_f, const _Float16* __restrict__ whh_b,
                       const float* __restrict__ bhh_f, const float* __restrict__ bhh_b,
                       float* __restrict__ h_f_out, float* __restrict__ h_b_out) {
  __shared__ _Float16 hbuf[16 * Hd];  // rows 8..15 stay zero (batch pad)
  int dir = blockIdx.x;
  const float*    xg  = dir ? xg_b : xg_f;
  const _Float16* whh = dir ? whh_b : whh_f;
  const float*    bhh = dir ? bhh_b : bhh_f;
  float*          hout = dir ? h_b_out : h_f_out;
  int tid = threadIdx.x;
  int wid = tid >> 5;
  int lane = tid & 31;
  int hi = lane >> 4, ln = lane & 15;
  int j0 = wid * 16;  // this wave's hidden-unit base

  for (int idx = tid; idx < 16 * Hd; idx += 1024) hbuf[idx] = (_Float16)0.f;
  __syncthreads();

  v8f cstate = {};  // c for (batch=r, unit=j0+ln), valid lanes hi==0
  for (int step = 0; step < Tt; ++step) {
    int t = dir ? (Tt - 1 - step) : step;
    v8f gi = {}, gf = {}, gg = {}, go = {};
    for (int k0 = 0; k0 < Hd; k0 += 32) {
      v16h a  = load_frag_a(hbuf, Hd, 0, k0);
      v16h b0 = load_frag_b(whh, Hd, 0 * Hd + j0, k0);
      v16h b1 = load_frag_b(whh, Hd, 1 * Hd + j0, k0);
      v16h b2 = load_frag_b(whh, Hd, 2 * Hd + j0, k0);
      v16h b3 = load_frag_b(whh, Hd, 3 * Hd + j0, k0);
      gi = WMMA_F16(a, b0, gi);
      gf = WMMA_F16(a, b1, gf);
      gg = WMMA_F16(a, b2, gg);
      go = WMMA_F16(a, b3, go);
    }
    __syncthreads();  // all waves done reading hbuf
    if (hi == 0) {
      int unit = j0 + ln;
#pragma unroll
      for (int r = 0; r < 8; ++r) {
        size_t base = ((size_t)r * Tt + t) * (4 * Hd);
        float iv = gi[r] + xg[base + unit]            + bhh[unit];
        float fv = gf[r] + xg[base + Hd + unit]       + bhh[Hd + unit];
        float gv = gg[r] + xg[base + 2 * Hd + unit]   + bhh[2 * Hd + unit];
        float ov = go[r] + xg[base + 3 * Hd + unit]   + bhh[3 * Hd + unit];
        iv = 1.f / (1.f + __expf(-iv));
        fv = 1.f / (1.f + __expf(-fv));
        gv = tanhf(gv);
        ov = 1.f / (1.f + __expf(-ov));
        float c = fv * cstate[r] + iv * gv;
        cstate[r] = c;
        float hval = ov * tanhf(c);
        hbuf[r * Hd + unit] = (_Float16)hval;
        hout[((size_t)r * Tt + t) * Hd + unit] = hval;
      }
    }
    __syncthreads();
  }
}

// ---------------- mean over T ----------------
__global__ void k_mean(const float* __restrict__ ln2, float* __restrict__ emb) {
  int i = blockIdx.x * 256 + threadIdx.x;  // 8 * 1024
  int b = i >> 10, j = i & 1023;
  float s = 0.f;
  for (int t = 0; t < Tt; ++t) s += ln2[((size_t)b * Tt + t) * (2 * Hd) + j];
  emb[i] = s * (1.f / (float)Tt);
}

// ---------------- final projection [8,1024] @ [512,1024]^T ----------------
__global__ void k_final(const float* __restrict__ emb, const float* __restrict__ w_out,
                        const float* __restrict__ b_out, float* __restrict__ out) {
  int i = blockIdx.x * 256 + threadIdx.x;  // 8 * 512
  int b = i >> 9, n = i & 511;
  float s = b_out[n];
  const float* e = emb + (size_t)b * (2 * Hd);
  const float* w = w_out + (size_t)n * (2 * Hd);
  for (int k = 0; k < 2 * Hd; ++k) s += e[k] * w[k];
  out[i] = s;
}

// ---------------- host orchestration ----------------
extern "C" void kernel_launch(void* const* d_in, const int* in_sizes, int n_in,
                              void* d_out, int out_size, void* d_ws, size_t ws_size,
                              hipStream_t stream) {
  const float* temporal = (const float*)d_in[0];
  const float* w_in  = (const float*)d_in[1];  const float* b_in  = (const float*)d_in[2];
  const float* wq    = (const float*)d_in[3];  const float* bq    = (const float*)d_in[4];
  const float* wk    = (const float*)d_in[5];  const float* bk    = (const float*)d_in[6];
  const float* wv    = (const float*)d_in[7];  const float* bv    = (const float*)d_in[8];
  const float* wo    = (const float*)d_in[9];  const float* bo    = (const float*)d_in[10];
  const float* g1    = (const float*)d_in[11]; const float* b1    = (const float*)d_in[12];
  const float* wih_f = (const float*)d_in[13]; const float* whh_f = (const float*)d_in[14];
  const float* bih_f = (const float*)d_in[15]; const float* bhh_f = (const float*)d_in[16];
  const float* wih_b = (const float*)d_in[17]; const float* whh_b = (const float*)d_in[18];
  const float* bih_b = (const float*)d_in[19]; const float* bhh_b = (const float*)d_in[20];
  const float* g2    = (const float*)d_in[21]; const float* b2    = (const float*)d_in[22];
  const float* w_out = (const float*)d_in[23]; const float* b_out = (const float*)d_in[24];
  float* out = (float*)d_out;

  char* base = (char*)d_ws;
  size_t off = 0;
  auto alloc = [&](size_t bytes) -> void* {
    void* p = base + off;
    off = (off + bytes + 255) & ~(size_t)255;
    return p;
  };

  _Float16* td_h    = (_Float16*)alloc((size_t)ROWS * D_IN * 2);
  _Float16* w_in_h  = (_Float16*)alloc((size_t)Hd * D_IN * 2);
  _Float16* wq_h    = (_Float16*)alloc((size_t)Hd * Hd * 2);
  _Float16* wk_h    = (_Float16*)alloc((size_t)Hd * Hd * 2);
  _Float16* wv_h    = (_Float16*)alloc((size_t)Hd * Hd * 2);
  _Float16* wo_h    = (_Float16*)alloc((size_t)Hd * Hd * 2);
  _Float16* wih_f_h = (_Float16*)alloc((size_t)4 * Hd * Hd * 2);
  _Float16* whh_f_h = (_Float16*)alloc((size_t)4 * Hd * Hd * 2);
  _Float16* wih_b_h = (_Float16*)alloc((size_t)4 * Hd * Hd * 2);
  _Float16* whh_b_h = (_Float16*)alloc((size_t)4 * Hd * Hd * 2);
  float*    x32     = (float*)   alloc((size_t)ROWS * Hd * 4);
  _Float16* x_h     = (_Float16*)alloc((size_t)ROWS * Hd * 2);
  _Float16* q_h     = (_Float16*)alloc((size_t)ROWS * Hd * 2);
  _Float16* k_h     = (_Float16*)alloc((size_t)ROWS * Hd * 2);
  _Float16* vt_h    = (_Float16*)alloc((size_t)ROWS * Hd * 2);
  _Float16* attn_h  = (_Float16*)alloc((size_t)ROWS * Hd * 2);
  float*    o32     = (float*)   alloc((size_t)ROWS * Hd * 4);
  float*    xln32   = (float*)   alloc((size_t)ROWS * Hd * 4);
  _Float16* xln_h   = (_Float16*)alloc((size_t)ROWS * Hd * 2);
  float*    xg_fb   = (float*)   alloc((size_t)ROWS * 4 * Hd * 4);
  float*    xg_bb   = (float*)   alloc((size_t)ROWS * 4 * Hd * 4);
  float*    hf      = (float*)   alloc((size_t)ROWS * Hd * 4);
  float*    hb      = (float*)   alloc((size_t)ROWS * Hd * 4);
  float*    ln2buf  = (float*)   alloc((size_t)ROWS * 2 * Hd * 4);
  float*    emb     = (float*)   alloc((size_t)Bb * 2 * Hd * 4);

  auto conv = [&](const float* s, _Float16* d, int n) {
    k_f32_to_f16<<<(n + 255) / 256, 256, 0, stream>>>(s, d, n);
  };
  conv(temporal, td_h, ROWS * D_IN);
  conv(w_in, w_in_h, Hd * D_IN);
  conv(wq, wq_h, Hd * Hd);  conv(wk, wk_h, Hd * Hd);
  conv(wv, wv_h, Hd * Hd);  conv(wo, wo_h, Hd * Hd);
  conv(wih_f, wih_f_h, 4 * Hd * Hd);  conv(whh_f, whh_f_h, 4 * Hd * Hd);
  conv(wih_b, wih_b_h, 4 * Hd * Hd);  conv(whh_b, whh_b_h, 4 * Hd * Hd);

  dim3 gblk(128);
  // x = temporal @ w_in^T + b_in + PE   -> f32 (residual) + f16 (GEMM input)
  k_gemm_wmma<<<dim3(Hd / 128, ROWS / 64), gblk, 0, stream>>>(
      td_h, w_in_h, b_in, x32, x_h, ROWS, Hd, D_IN, GF_PE);
  // Q, K, V projections
  k_gemm_wmma<<<dim3(Hd / 128, ROWS / 64), gblk, 0, stream>>>(
      x_h, wq_h, bq, nullptr, q_h, ROWS, Hd, Hd, 0);
  k_gemm_wmma<<<dim3(Hd / 128, ROWS / 64), gblk, 0, stream>>>(
      x_h, wk_h, bk, nullptr, k_h, ROWS, Hd, Hd, 0);
  k_gemm_wmma<<<dim3(Hd / 128, ROWS / 64), gblk, 0, stream>>>(
      x_h, wv_h, bv, nullptr, vt_h, ROWS, Hd, Hd, GF_VT);
  // flash causal attention
  k_flash_attn<<<dim3(Tt / 16, Bb * NH), dim3(32), 0, stream>>>(q_h, k_h, vt_h, attn_h);
  // output projection
  k_gemm_wmma<<<dim3(Hd / 128, ROWS / 64), gblk, 0, stream>>>(
      attn_h, wo_h, bo, o32, nullptr, ROWS, Hd, Hd, 0);
  // LN1 with residual
  k_layernorm<<<ROWS, 256, 0, stream>>>(o32, nullptr, x32, g1, b1, xln32, xln_h, Hd, 0);
  // LSTM input gates (both directions)
  k_gemm_wmma<<<dim3(4 * Hd / 128, ROWS / 64), gblk, 0, stream>>>(
      xln_h, wih_f_h, bih_f, xg_fb, nullptr, ROWS, 4 * Hd, Hd, 0);
  k_gemm_wmma<<<dim3(4 * Hd / 128, ROWS / 64), gblk, 0, stream>>>(
      xln_h, wih_b_h, bih_b, xg_bb, nullptr, ROWS, 4 * Hd, Hd, 0);
  // persistent biLSTM recurrence
  k_lstm<<<2, 1024, 0, stream>>>(xg_fb, xg_bb, whh_f_h, whh_b_h, bhh_f, bhh_b, hf, hb);
  // LN2 over concat
  k_layernorm<<<ROWS, 256, 0, stream>>>(hf, hb, nullptr, g2, b2, ln2buf, nullptr, Hd, Hd);
  // mean over T, final projection
  k_mean<<<(Bb * 2 * Hd) / 256, 256, 0, stream>>>(ln2buf, emb);
  k_final<<<(Bb * Hd) / 256, 256, 0, stream>>>(emb, w_out, b_out, out);
}
